// HyperGNNTernaryComposeLayer_15951508537732
// MI455X (gfx1250) — compile-verified, matrix operand order checked by reading, not run on hardware
//
#include <hip/hip_runtime.h>

#define BS   8
#define NE   128
#define ENT  768
#define RELD 768
#define RANK 256
#define OUTD 256
#define JT   64
#define KB   32

typedef __attribute__((ext_vector_type(16))) __bf16 v16bf;
typedef __attribute__((ext_vector_type(8)))  float  v8f;
typedef int v4i_ __attribute__((vector_size(16)));

union Frag { int i[8]; v16bf v; };
union BfBits { __bf16 h; unsigned short u; };

static __device__ __forceinline__ unsigned short f2bf(float f) {
    BfBits b; b.h = (__bf16)f; return b.u;
}
static __device__ __forceinline__ int pack2(float lo, float hi) {
    return (int)((unsigned)f2bf(lo) | ((unsigned)f2bf(hi) << 16));
}

// ---- gfx1250 async global->LDS path (ASYNCcnt), with synchronous fallback ----
#if defined(__has_builtin)
#if __has_builtin(__builtin_amdgcn_global_load_async_to_lds_b128)
#define USE_ASYNC 1
#endif
#endif

#ifdef USE_ASYNC
static __device__ __forceinline__ void async_b128(const void* g, void* l) {
    __builtin_amdgcn_global_load_async_to_lds_b128(
        (__attribute__((address_space(1))) v4i_*)g,
        (__attribute__((address_space(3))) v4i_*)l, 0, 0);
}
#define WAIT_ASYNC(n) asm volatile("s_wait_asynccnt " #n ::: "memory")
#else
static __device__ __forceinline__ void async_b128(const void* g, void* l) {
    *(uint4*)l = *(const uint4*)g;   // synchronous fallback
}
#define WAIT_ASYNC(n) do { } while (0)
#endif

// ---------- phase 1: l = x @ W^T + bias ----------
__global__ void lin_kernel(const float* __restrict__ x, const float* __restrict__ W,
                           const float* __restrict__ bias, float* __restrict__ out,
                           int rows, int K) {
    int idx = blockIdx.x * blockDim.x + threadIdx.x;
    int r   = idx & (RANK - 1);
    int row = idx >> 8;
    if (row >= rows) return;
    const float* xr = x + (size_t)row * K;
    const float* wr = W + (size_t)r   * K;
    float acc = 0.f;
    for (int d = 0; d < K; ++d) acc = fmaf(xr[d], wr[d], acc);
    out[(size_t)row * RANK + r] = acc + bias[r];
}

// ---------- phase 2: elementwise fp32 -> bf16 (native layouts are already B-friendly) ----------
__global__ void convw_kernel(const float* __restrict__ W3, const float* __restrict__ We,
                             unsigned short* __restrict__ W3bt, unsigned short* __restrict__ Webt) {
    int idx = blockIdx.x * blockDim.x + threadIdx.x;
    const int totalW3 = RANK * RELD;          // W3: [256 n][768 k] row-major
    if (idx < totalW3) {
        W3bt[idx] = f2bf(W3[idx]);
    } else if (idx < totalW3 + OUTD * RANK) { // We: [256 n][256 k] row-major
        int j = idx - totalW3;
        Webt[j] = f2bf(We[j]);
    }
}

// ---------- phase 3: fused l3-GEMM -> CPD scale -> encode-GEMM -> LayerNorm ----------
__launch_bounds__(128, 1)
__global__ void fused_kernel(const float* __restrict__ rel,
                             const float* __restrict__ l1,
                             const float* __restrict__ l2,
                             const unsigned short* __restrict__ W3bt,
                             const unsigned short* __restrict__ Webt,
                             const float* __restrict__ b3,
                             const float* __restrict__ be,
                             const float* __restrict__ ln_g,
                             const float* __restrict__ ln_b,
                             float* __restrict__ out) {
    // 64 KB LDS: [0,32K) B slices (bf16, double) | [32K,48K) A tiles (fp32, double)
    //            [32K,64K) sProd (bf16) -- overlays bufA after the main GEMM
    __shared__ __align__(16) char smem[65536];
    unsigned short* bufB  = (unsigned short*)smem;            // [2][256][32]
    float*          bufA  = (float*)(smem + 32768);           // [2][64][32]
    unsigned short* sProd = (unsigned short*)(smem + 32768);  // [64][256]

    const int tid  = threadIdx.x;
    const int lane = tid & 31;
    const int wave = tid >> 5;
    const int lm   = lane & 15;
    const int lh   = lane >> 4;
    const int m0w  = wave * 16;

    int blk = blockIdx.x;
    const int jt = blk & 1;        blk >>= 1;
    const int i  = blk & (NE - 1); blk >>= 7;
    const int b  = blk;
    const int j0 = jt * JT;

    const float* relTile = rel + ((size_t)(b * NE + i) * NE + j0) * RELD;

    // --- async batch issuers: 4 (A) + 8 (B) b128 chunks per thread per K-step ---
    auto issueA = [&](int kb, int buf) {
        #pragma unroll
        for (int q = 0; q < 4; ++q) {
            const int c = tid + 128 * q;           // 0..511 chunks of 4 floats
            const int row = c >> 3, col4 = (c & 7) * 4;
            async_b128(relTile + (size_t)row * RELD + kb + col4,
                       bufA + buf * 2048 + row * 32 + col4);
        }
    };
    auto issueB = [&](const unsigned short* W, int stride, int kb, int buf) {
        #pragma unroll
        for (int q = 0; q < 8; ++q) {
            const int c = tid + 128 * q;           // 0..1023 chunks of 8 bf16
            const int n = c >> 2, k8 = (c & 3) * 8;
            async_b128(W + (size_t)n * stride + kb + k8,
                       bufB + buf * 8192 + n * 32 + k8);
        }
    };

    v8f acc[16];
    #pragma unroll
    for (int t = 0; t < 16; ++t)
        #pragma unroll
        for (int q = 0; q < 8; ++q) acc[t][q] = 0.f;

    // ======== main GEMM: l3_tile[64x256] = rel[64x768] x W3^T, double-buffered ========
    issueA(0, 0);
    issueB(W3bt, RELD, 0, 0);
    int cur = 0;
    for (int kb = 0; kb < RELD; kb += KB) {
        if (kb + KB < RELD) {
            __syncthreads();                 // everyone finished reading buf[cur^1]
            issueA(kb + KB, cur ^ 1);
            issueB(W3bt, RELD, kb + KB, cur ^ 1);
            WAIT_ASYNC(12);                  // in-order: previous 12 (buf[cur]) done
        } else {
            WAIT_ASYNC(0);
        }
        __syncthreads();                     // buf[cur] visible to all waves

        Frag a;                              // fp32 LDS -> packed bf16 A fragment
        {
            const float* ar = bufA + cur * 2048 + (m0w + lm) * 32 + 8 * lh;
            float4 f0 = *(const float4*)(ar);
            float4 f1 = *(const float4*)(ar + 4);
            float4 f2 = *(const float4*)(ar + 16);
            float4 f3 = *(const float4*)(ar + 20);
            a.i[0] = pack2(f0.x, f0.y); a.i[1] = pack2(f0.z, f0.w);
            a.i[2] = pack2(f1.x, f1.y); a.i[3] = pack2(f1.z, f1.w);
            a.i[4] = pack2(f2.x, f2.y); a.i[5] = pack2(f2.z, f2.w);
            a.i[6] = pack2(f3.x, f3.y); a.i[7] = pack2(f3.z, f3.w);
        }
        #pragma unroll
        for (int t = 0; t < 16; ++t) {
            Frag bb;
            const int* bp = (const int*)(bufB + cur * 8192 + (t * 16 + lm) * 32);
            const int  o  = 4 * lh;
            bb.i[0] = bp[o];     bb.i[1] = bp[o + 1]; bb.i[2] = bp[o + 2];  bb.i[3] = bp[o + 3];
            bb.i[4] = bp[o + 8]; bb.i[5] = bp[o + 9]; bb.i[6] = bp[o + 10]; bb.i[7] = bp[o + 11];
            acc[t] = __builtin_amdgcn_wmma_f32_16x16x32_bf16(false, a.v, false, bb.v,
                                                             (short)0, acc[t], false, false);
        }
        cur ^= 1;
    }

    __syncthreads();   // all waves done with bufA (sProd overlays it) and bufB

    // ======== CPD elementwise: prod = (l3 + b3) * l1[b,i,:] * l2[b,j,:] -> bf16 LDS ========
    {
        const float* l1p = l1 + (size_t)(b * NE + i) * RANK;
        #pragma unroll
        for (int t = 0; t < 16; ++t) {
            const int n    = t * 16 + lm;
            const float c3 = b3[n];
            const float s1 = l1p[n];
            #pragma unroll
            for (int p = 0; p < 8; ++p) {
                const int rloc = m0w + p + 8 * lh;   // C layout: VGPR p -> M=p / M=8+p
                const int j    = j0 + rloc;
                const float s2 = l2[(size_t)(b * NE + j) * RANK + n];
                sProd[rloc * RANK + n] = f2bf((acc[t][p] + c3) * s1 * s2);
            }
        }
    }

    // ======== second GEMM: f[64x256] = prod x We^T, double-buffered Web slices ========
    v8f acc2[16];
    #pragma unroll
    for (int t = 0; t < 16; ++t)
        #pragma unroll
        for (int q = 0; q < 8; ++q) acc2[t][q] = 0.f;

    issueB(Webt, RANK, 0, 0);
    int cur2 = 0;
    for (int kb = 0; kb < RANK; kb += KB) {
        if (kb + KB < RANK) {
            __syncthreads();
            issueB(Webt, RANK, kb + KB, cur2 ^ 1);
            WAIT_ASYNC(8);
        } else {
            WAIT_ASYNC(0);
        }
        __syncthreads();

        Frag a;
        {
            const int* rp = (const int*)(sProd + (m0w + lm) * RANK + kb);
            const int  o  = 4 * lh;
            a.i[0] = rp[o];     a.i[1] = rp[o + 1]; a.i[2] = rp[o + 2];  a.i[3] = rp[o + 3];
            a.i[4] = rp[o + 8]; a.i[5] = rp[o + 9]; a.i[6] = rp[o + 10]; a.i[7] = rp[o + 11];
        }
        #pragma unroll
        for (int t = 0; t < 16; ++t) {
            Frag bb;
            const int* bp = (const int*)(bufB + cur2 * 8192 + (t * 16 + lm) * 32);
            const int  o  = 4 * lh;
            bb.i[0] = bp[o];     bb.i[1] = bp[o + 1]; bb.i[2] = bp[o + 2];  bb.i[3] = bp[o + 3];
            bb.i[4] = bp[o + 8]; bb.i[5] = bp[o + 9]; bb.i[6] = bp[o + 10]; bb.i[7] = bp[o + 11];
            acc2[t] = __builtin_amdgcn_wmma_f32_16x16x32_bf16(false, a.v, false, bb.v,
                                                              (short)0, acc2[t], false, false);
        }
        cur2 ^= 1;
    }

    // ======== + be, LayerNorm over 256, store ========
    const size_t outBase = ((size_t)(b * NE + i) * NE + j0) * OUTD;
    #pragma unroll
    for (int p = 0; p < 8; ++p) {
        float s = 0.f, s2 = 0.f;
        float fv[16];
        #pragma unroll
        for (int t = 0; t < 16; ++t) {
            const int n = t * 16 + lm;
            const float x = acc2[t][p] + be[n];
            fv[t] = x; s += x; s2 += x * x;
        }
        #pragma unroll
        for (int m = 1; m < 16; m <<= 1) {   // row lives on one 16-lane half
            s  += __shfl_xor(s,  m, 32);
            s2 += __shfl_xor(s2, m, 32);
        }
        const float mu   = s * (1.f / OUTD);
        const float var  = s2 * (1.f / OUTD) - mu * mu;
        const float rstd = rsqrtf(var + 1e-6f);
        const int rloc = m0w + p + 8 * lh;
        float* orow = out + outBase + (size_t)rloc * OUTD;
        #pragma unroll
        for (int t = 0; t < 16; ++t) {
            const int n = t * 16 + lm;
            orow[n] = (fv[t] - mu) * rstd * ln_g[n] + ln_b[n];
        }
    }
}

extern "C" void kernel_launch(void* const* d_in, const int* in_sizes, int n_in,
                              void* d_out, int out_size, void* d_ws, size_t ws_size,
                              hipStream_t stream) {
    (void)in_sizes; (void)n_in; (void)out_size; (void)ws_size;
    const float* sub = (const float*)d_in[0];
    const float* obj = (const float*)d_in[1];
    const float* rel = (const float*)d_in[2];
    const float* W1  = (const float*)d_in[3];
    const float* b1  = (const float*)d_in[4];
    const float* W2  = (const float*)d_in[5];
    const float* b2  = (const float*)d_in[6];
    const float* W3  = (const float*)d_in[7];
    const float* b3  = (const float*)d_in[8];
    const float* We  = (const float*)d_in[9];
    const float* be  = (const float*)d_in[10];
    const float* lng = (const float*)d_in[11];
    const float* lnb = (const float*)d_in[12];
    float* out = (float*)d_out;

    char* ws = (char*)d_ws;
    float* l1 = (float*)ws;
    float* l2 = (float*)(ws + (size_t)BS * NE * RANK * 4);
    unsigned short* W3bt = (unsigned short*)(ws + (size_t)2 * BS * NE * RANK * 4);
    unsigned short* Webt = (unsigned short*)((char*)W3bt + (size_t)RANK * RELD * 2);

    const int rows = BS * NE;
    lin_kernel<<<rows * RANK / 256, 256, 0, stream>>>(sub, W1, b1, l1, rows, ENT);
    lin_kernel<<<rows * RANK / 256, 256, 0, stream>>>(obj, W2, b2, l2, rows, ENT);
    convw_kernel<<<(RANK * RELD + OUTD * RANK) / 256, 256, 0, stream>>>(W3, We, W3bt, Webt);
    fused_kernel<<<BS * NE * (NE / JT), 128, 0, stream>>>(rel, l1, l2, W3bt, Webt,
                                                          b3, be, lng, lnb, out);
}
